// TransConvLayer_592705487199
// MI455X (gfx1250) — matrix operand order repaired and checked by reading, not compiled, not run on hardware
//
#include <hip/hip_runtime.h>
#include <hip/hip_bf16.h>

// ---------------- problem constants ----------------
#define NROWS   100000
#define INCH    256
#define HD      256      // HEADS * OUT_CH
#define OUTCH   64
#define HEADS   4
#define ROWTILES ((NROWS + 63) / 64)   // 1563
#define KV_CHUNK 1024
#define KV_BLKS  ((NROWS + KV_CHUNK - 1) / KV_CHUNK)  // 98

// ---------------- vector typedefs ----------------
typedef __attribute__((ext_vector_type(16))) __bf16          v16bf;
typedef __attribute__((ext_vector_type(16))) unsigned short  v16u;
typedef __attribute__((ext_vector_type(8)))  unsigned short  u8u;
typedef __attribute__((ext_vector_type(4)))  unsigned short  u4u;
typedef __attribute__((ext_vector_type(8)))  float           v8f;
typedef __attribute__((ext_vector_type(4)))  float           f4;

// ---------------- helpers (native bf16 converts) ----------------
__device__ __forceinline__ unsigned short f2bf(float f) {
  __bf16 b = (__bf16)f;                       // v_cvt_*_bf16_f32
  return __builtin_bit_cast(unsigned short, b);
}
__device__ __forceinline__ float bf2f(unsigned short u) {
  return (float)__builtin_bit_cast(__bf16, u); // shift / cvt
}
__device__ __forceinline__ int imin(int a, int b) { return a < b ? a : b; }

__device__ __forceinline__ v8f wmma_bf16(v16u a, v16u b, v8f c) {
  return __builtin_amdgcn_wmma_f32_16x16x32_bf16(
      false, __builtin_bit_cast(v16bf, a),
      false, __builtin_bit_cast(v16bf, b),
      (short)0, c, false, false);
}

// =====================================================================
// Kernel 0: zero f32 accumulators (kv[4*64*64], ksum[256], sumsq[2])
// =====================================================================
__global__ __launch_bounds__(256) void zero_kernel(float* p, int n) {
  int i = blockIdx.x * 256 + threadIdx.x;
  if (i < n) p[i] = 0.0f;
}

// =====================================================================
// Kernel 0b: convert a weight matrix fp32 -> bf16 (done once; L2-hot after)
// =====================================================================
__global__ __launch_bounds__(256) void cvtw_kernel(const float* __restrict__ src,
                                                   unsigned short* __restrict__ dst,
                                                   int n) {
  int i4 = (blockIdx.x * 256 + threadIdx.x) * 4;
  if (i4 < n) {
    f4 v = ((const f4*)(src + i4))[0];
    u4u o;
#pragma unroll
    for (int i = 0; i < 4; i++) o[i] = f2bf(v[i]);
    ((u4u*)(dst + i4))[0] = o;
  }
}

// =====================================================================
// Kernel 1: q/k/v projection.  grid = ROWTILES; block = 256 (8 waves).
// Stage Xq/Xs 64x256 row-slabs once into LDS as bf16, then sweep all
// 12 output col-tiles (4 q, 4 k, 4 v) with WMMA; weights are bf16 in L2.
// =====================================================================
#define XPITCH 264   // halves per LDS row (528B: 16B aligned, bank-staggered)
__global__ __launch_bounds__(256) void proj_kernel(
    const float* __restrict__ Xq, const float* __restrict__ Xs,
    const unsigned short* __restrict__ wqb, const unsigned short* __restrict__ wkb,
    const unsigned short* __restrict__ wvb,
    const float* __restrict__ bq, const float* __restrict__ bk,
    const float* __restrict__ bv,
    unsigned short* __restrict__ qws, unsigned short* __restrict__ kws,
    unsigned short* __restrict__ vws, float* __restrict__ sumsq) {
  __shared__ unsigned int xqs[64 * (XPITCH / 2)];
  __shared__ unsigned int xss[64 * (XPITCH / 2)];
  __shared__ float red[256];

  const int tid  = threadIdx.x;
  const int lane = tid & 31;
  const int w    = tid >> 5;
  const int m_t  = w & 3;            // 16-row subtile
  const int d_t0 = (w >> 2) * 2;     // first of two 16-col subtiles
  const int lrow = lane & 15;
  const int hi   = lane >> 4;        // lane half
  const int r0   = blockIdx.x * 64;

  // ---- stage: thread t loads 64 consecutive floats of row t/4 for both X's
  {
    const int srow = tid >> 2;
    const int cb   = (tid & 3) * 64;
    const int gr   = imin(r0 + srow, NROWS - 1);
    const float* xqp = Xq + (size_t)gr * INCH + cb;
    const float* xsp = Xs + (size_t)gr * INCH + cb;
    unsigned int* dq = &xqs[srow * (XPITCH / 2) + cb / 2];
    unsigned int* ds = &xss[srow * (XPITCH / 2) + cb / 2];
#pragma unroll
    for (int j = 0; j < 64; j += 4) {
      f4 a = ((const f4*)(xqp + j))[0];
      f4 b = ((const f4*)(xsp + j))[0];
      dq[j / 2]     = (unsigned int)f2bf(a[0]) | ((unsigned int)f2bf(a[1]) << 16);
      dq[j / 2 + 1] = (unsigned int)f2bf(a[2]) | ((unsigned int)f2bf(a[3]) << 16);
      ds[j / 2]     = (unsigned int)f2bf(b[0]) | ((unsigned int)f2bf(b[1]) << 16);
      ds[j / 2 + 1] = (unsigned int)f2bf(b[2]) | ((unsigned int)f2bf(b[3]) << 16);
    }
  }
  __syncthreads();

  float sq_q = 0.0f, sq_k = 0.0f;
  const int arow = m_t * 16 + lrow;        // LDS row for A fragment
  const unsigned short* xq16 = (const unsigned short*)xqs;
  const unsigned short* xs16 = (const unsigned short*)xss;

#pragma unroll
  for (int ct = 0; ct < 12; ct++) {
    const int which   = ct >> 2;           // 0=q, 1=k, 2=v
    const int colbase = (ct & 3) * 64;
    const unsigned short* Wb = (which == 0) ? wqb : (which == 1) ? wkb : wvb;
    const float* B           = (which == 0) ? bq  : (which == 1) ? bk  : bv;
    const unsigned short* xt = (which == 0) ? xq16 : xs16;
    unsigned short* dst      = (which == 0) ? qws : (which == 1) ? kws : vws;

    const int o0 = colbase + d_t0 * 16 + lrow;
    const int o1 = o0 + 16;
    const unsigned short* b0p = Wb + (size_t)o0 * INCH + hi * 16;
    const unsigned short* b1p = Wb + (size_t)o1 * INCH + hi * 16;

    v8f acc0 = {}, acc1 = {};
#pragma unroll
    for (int c0 = 0; c0 < INCH; c0 += 32) {
      // A fragment from LDS: lanes<16 K {0..15}, lanes>=16 {16..31}
      const unsigned short* ap = xt + arow * XPITCH + c0 + hi * 16;
      u8u a0 = ((const u8u*)ap)[0];
      u8u a1 = ((const u8u*)(ap + 8))[0];
      v16u au;
#pragma unroll
      for (int i = 0; i < 8; i++) { au[i] = a0[i]; au[8 + i] = a1[i]; }
      // B fragments from bf16 weights (L2-hot): 16 contiguous halves
      u8u b00 = ((const u8u*)(b0p + c0))[0];
      u8u b01 = ((const u8u*)(b0p + c0 + 8))[0];
      u8u b10 = ((const u8u*)(b1p + c0))[0];
      u8u b11 = ((const u8u*)(b1p + c0 + 8))[0];
      v16u bu0, bu1;
#pragma unroll
      for (int i = 0; i < 8; i++) {
        bu0[i] = b00[i]; bu0[8 + i] = b01[i];
        bu1[i] = b10[i]; bu1[8 + i] = b11[i];
      }
      acc0 = wmma_bf16(au, bu0, acc0);
      acc1 = wmma_bf16(au, bu1, acc1);
    }

    // bias, sumsq, bf16 store
    const float bb0 = B[o0], bb1 = B[o1];
    float loc = 0.0f;
#pragma unroll
    for (int g = 0; g < 8; g++) {
      acc0[g] += bb0; acc1[g] += bb1;
      const int rr = r0 + m_t * 16 + g + hi * 8;
      if (rr < NROWS) {
        dst[(size_t)rr * HD + o0] = f2bf(acc0[g]);
        dst[(size_t)rr * HD + o1] = f2bf(acc1[g]);
        loc += acc0[g] * acc0[g] + acc1[g] * acc1[g];
      }
    }
    if (which == 0) sq_q += loc;
    else if (which == 1) sq_k += loc;
  }

  // block reductions for the two global norms
  __syncthreads();
  red[tid] = sq_q;
  __syncthreads();
#pragma unroll
  for (int s = 128; s > 0; s >>= 1) {
    if (tid < s) red[tid] += red[tid + s];
    __syncthreads();
  }
  if (tid == 0) atomicAdd(&sumsq[0], red[0]);
  __syncthreads();
  red[tid] = sq_k;
  __syncthreads();
#pragma unroll
  for (int s = 128; s > 0; s >>= 1) {
    if (tid < s) red[tid] += red[tid + s];
    __syncthreads();
  }
  if (tid == 0) atomicAdd(&sumsq[1], red[0]);
}

// =====================================================================
// Kernel 2: kv[h] = K_h^T V_h  (64x64 per head, K-dim = N), + ksum
// grid = (KV_BLKS, HEADS); block = 256
// =====================================================================
#define P3 40   // LDS pitch in halves (80B: 16B aligned, conflict-friendly)
__global__ __launch_bounds__(256) void kv_kernel(
    const unsigned short* __restrict__ kws,
    const unsigned short* __restrict__ vws,
    float* __restrict__ kv, float* __restrict__ ksum) {
  __shared__ unsigned short kT[64 * P3];
  __shared__ unsigned short vT[64 * P3];

  const int tid  = threadIdx.x;
  const int lane = tid & 31;
  const int w    = tid >> 5;
  const int m_t  = w & 3;
  const int d_t0 = (w >> 2) * 2;
  const int lrow = lane & 15;
  const int hi   = lane >> 4;
  const int h    = blockIdx.y;
  const int nbase = blockIdx.x * KV_CHUNK;

  const int nl = tid >> 3;        // 0..31  (local row)
  const int fb = (tid & 7) * 8;   // feature block base

  v8f acc0 = {}, acc1 = {};
  float ks_loc = 0.0f;

  for (int st = 0; st < KV_CHUNK; st += 32) {
    const int row = nbase + st + nl;
    u8u rk = {}, rv = {};
    if (row < NROWS) {
      rk = ((const u8u*)(kws + (size_t)row * HD + h * 64 + fb))[0];
      rv = ((const u8u*)(vws + (size_t)row * HD + h * 64 + fb))[0];
    }
    __syncthreads();   // prior consumers done
#pragma unroll
    for (int j = 0; j < 8; j++) {     // transposed store: [feature][row]
      kT[(fb + j) * P3 + nl] = rk[j];
      vT[(fb + j) * P3 + nl] = rv[j];
    }
    __syncthreads();

    // ksum partial (threads 0..63, feature = tid)
    if (tid < 64) {
#pragma unroll
      for (int j = 0; j < 4; j++) {
        u8u t = ((const u8u*)(&kT[tid * P3 + j * 8]))[0];
#pragma unroll
        for (int i = 0; i < 8; i++) ks_loc += bf2f(t[i]);
      }
    }

    // A = K^T tile (contiguous in n after the transpose), B = V tile
    const int asel = hi * 16;
    u8u a0 = ((const u8u*)(&kT[(m_t * 16 + lrow) * P3 + asel]))[0];
    u8u a1 = ((const u8u*)(&kT[(m_t * 16 + lrow) * P3 + asel + 8]))[0];
    v16u au;
#pragma unroll
    for (int i = 0; i < 8; i++) { au[i] = a0[i]; au[8 + i] = a1[i]; }

    const int bsel = hi * 16;
    u8u b00 = ((const u8u*)(&vT[(d_t0 * 16 + lrow) * P3 + bsel]))[0];
    u8u b01 = ((const u8u*)(&vT[(d_t0 * 16 + lrow) * P3 + bsel + 8]))[0];
    u8u b10 = ((const u8u*)(&vT[((d_t0 + 1) * 16 + lrow) * P3 + bsel]))[0];
    u8u b11 = ((const u8u*)(&vT[((d_t0 + 1) * 16 + lrow) * P3 + bsel + 8]))[0];
    v16u bu0, bu1;
#pragma unroll
    for (int i = 0; i < 8; i++) {
      bu0[i] = b00[i]; bu0[8 + i] = b01[i];
      bu1[i] = b10[i]; bu1[8 + i] = b11[i];
    }
    acc0 = wmma_bf16(au, bu0, acc0);
    acc1 = wmma_bf16(au, bu1, acc1);
  }

#pragma unroll
  for (int g = 0; g < 8; g++) {
    const int mm = m_t * 16 + g + hi * 8;
    atomicAdd(&kv[((size_t)h * 64 + mm) * 64 + d_t0 * 16 + lrow], acc0[g]);
    atomicAdd(&kv[((size_t)h * 64 + mm) * 64 + (d_t0 + 1) * 16 + lrow], acc1[g]);
  }
  if (tid < 64) atomicAdd(&ksum[h * 64 + tid], ks_loc);
}

// =====================================================================
// Kernel 3: num = Q_h x kv_h, den = Q_h . ksum_h, combine + head mean
// grid = ROWTILES; block = 256
// =====================================================================
#define P4 72   // LDS pitch in halves for kvT (144B rows)
__global__ __launch_bounds__(256) void out_kernel(
    const unsigned short* __restrict__ qws,
    const unsigned short* __restrict__ vws,
    const float* __restrict__ kv, const float* __restrict__ ksum,
    const float* __restrict__ sumsq, float* __restrict__ out) {
  __shared__ unsigned short kvT[64 * P4];
  __shared__ float ksm[64];
  __shared__ float denl[64];

  const int tid  = threadIdx.x;
  const int lane = tid & 31;
  const int w    = tid >> 5;
  const int m_t  = w & 3;
  const int d_t0 = (w >> 2) * 2;
  const int lrow = lane & 15;
  const int hi   = lane >> 4;
  const int r0   = blockIdx.x * 64;

  const float nq  = sqrtf(sumsq[0]);
  const float nk  = sqrtf(sumsq[1]);
  const float inv = 1.0f / (nq * nk);
  const float Nf  = (float)NROWS;

  v8f oacc0 = {}, oacc1 = {};

  for (int h = 0; h < HEADS; h++) {
    __syncthreads();   // previous head's LDS reads done
    // stage kv_h transposed as bf16: kvT[d][m]
    {
      const int m  = tid >> 2;
      const int d0 = (tid & 3) * 16;
      const float* src = kv + ((size_t)h * 64 + m) * 64 + d0;
#pragma unroll
      for (int j = 0; j < 16; j++) kvT[(d0 + j) * P4 + m] = f2bf(src[j]);
    }
    if (tid < 64) ksm[tid] = ksum[h * 64 + tid];
    __syncthreads();

    // normalizer GEMV: den[r] = q[r,h,:] . ksum_h
    if (tid < 64) {
      const int rc = imin(r0 + tid, NROWS - 1);
      const u8u* qp = (const u8u*)(qws + (size_t)rc * HD + h * 64);
      float d = 0.0f;
#pragma unroll
      for (int j = 0; j < 8; j++) {
        u8u t = qp[j];
#pragma unroll
        for (int i = 0; i < 8; i++) d += bf2f(t[i]) * ksm[j * 8 + i];
      }
      denl[tid] = d;
    }
    __syncthreads();

    // num = Q_h (64x64) x kv_h (64x64), two 32-deep WMMA steps
    v8f n0 = {}, n1 = {};
#pragma unroll
    for (int c0 = 0; c0 < 64; c0 += 32) {
      const int rc = imin(r0 + m_t * 16 + lrow, NROWS - 1);
      const unsigned short* qrow = qws + (size_t)rc * HD + h * 64 + c0 + hi * 16;
      u8u a0 = ((const u8u*)qrow)[0];
      u8u a1 = ((const u8u*)(qrow + 8))[0];
      v16u au;
#pragma unroll
      for (int i = 0; i < 8; i++) { au[i] = a0[i]; au[8 + i] = a1[i]; }

      const int bsel = c0 + hi * 16;
      u8u b00 = ((const u8u*)(&kvT[(d_t0 * 16 + lrow) * P4 + bsel]))[0];
      u8u b01 = ((const u8u*)(&kvT[(d_t0 * 16 + lrow) * P4 + bsel + 8]))[0];
      u8u b10 = ((const u8u*)(&kvT[((d_t0 + 1) * 16 + lrow) * P4 + bsel]))[0];
      u8u b11 = ((const u8u*)(&kvT[((d_t0 + 1) * 16 + lrow) * P4 + bsel + 8]))[0];
      v16u bu0, bu1;
#pragma unroll
      for (int i = 0; i < 8; i++) {
        bu0[i] = b00[i]; bu0[8 + i] = b01[i];
        bu1[i] = b10[i]; bu1[8 + i] = b11[i];
      }
      n0 = wmma_bf16(au, bu0, n0);
      n1 = wmma_bf16(au, bu1, n1);
    }

    // combine: (num*inv + N*v) / (den*inv + N), accumulate over heads
#pragma unroll
    for (int g = 0; g < 8; g++) {
      const int rloc = m_t * 16 + g + hi * 8;
      const int rc = imin(r0 + rloc, NROWS - 1);
      const float den = denl[rloc] * inv + Nf;
      const int d0 = d_t0 * 16 + lrow, d1 = d0 + 16;
      const float v0 = bf2f(vws[(size_t)rc * HD + h * 64 + d0]);
      const float v1 = bf2f(vws[(size_t)rc * HD + h * 64 + d1]);
      oacc0[g] += (n0[g] * inv + Nf * v0) / den;
      oacc1[g] += (n1[g] * inv + Nf * v1) / den;
    }
  }

#pragma unroll
  for (int g = 0; g < 8; g++) {
    const int r = r0 + m_t * 16 + g + hi * 8;
    if (r < NROWS) {
      out[(size_t)r * OUTCH + d_t0 * 16 + lrow]        = oacc0[g] * 0.25f;
      out[(size_t)r * OUTCH + (d_t0 + 1) * 16 + lrow]  = oacc1[g] * 0.25f;
    }
  }
}

// =====================================================================
// host launcher
// =====================================================================
extern "C" void kernel_launch(void* const* d_in, const int* in_sizes, int n_in,
                              void* d_out, int out_size, void* d_ws, size_t ws_size,
                              hipStream_t stream) {
  const float* Xq = (const float*)d_in[0];
  const float* Xs = (const float*)d_in[1];
  const float* Wq = (const float*)d_in[2];
  const float* bq = (const float*)d_in[3];
  const float* Wk = (const float*)d_in[4];
  const float* bk = (const float*)d_in[5];
  const float* Wv = (const float*)d_in[6];
  const float* bv = (const float*)d_in[7];
  float* out = (float*)d_out;

  // workspace layout
  unsigned short* qws = (unsigned short*)d_ws;
  unsigned short* kws = qws + (size_t)NROWS * HD;
  unsigned short* vws = kws + (size_t)NROWS * HD;
  float* kv    = (float*)(vws + (size_t)NROWS * HD);
  float* ksum  = kv + HEADS * 64 * 64;
  float* sumsq = ksum + HD;
  unsigned short* wqb = (unsigned short*)(sumsq + 2);
  unsigned short* wkb = wqb + INCH * HD;
  unsigned short* wvb = wkb + INCH * HD;

  const int nzero = HEADS * 64 * 64 + HD + 2;
  zero_kernel<<<(nzero + 255) / 256, 256, 0, stream>>>(kv, nzero);

  const int wn = INCH * HD;  // 65536
  cvtw_kernel<<<wn / 1024, 256, 0, stream>>>(Wq, wqb, wn);
  cvtw_kernel<<<wn / 1024, 256, 0, stream>>>(Wk, wkb, wn);
  cvtw_kernel<<<wn / 1024, 256, 0, stream>>>(Wv, wvb, wn);

  proj_kernel<<<ROWTILES, 256, 0, stream>>>(
      Xq, Xs, wqb, wkb, wvb, bq, bk, bv, qws, kws, vws, sumsq);

  kv_kernel<<<dim3(KV_BLKS, HEADS), 256, 0, stream>>>(kws, vws, kv, ksum);

  out_kernel<<<ROWTILES, 256, 0, stream>>>(qws, vws, kv, ksum, sumsq, out);
}
